// SSIMLoss_6279242187142
// MI455X (gfx1250) — matrix operand order repaired and checked by reading, not compiled, not run on hardware
//
#include <hip/hip_runtime.h>

typedef __attribute__((ext_vector_type(16))) _Float16     v16h;
typedef __attribute__((ext_vector_type(8)))  float        v8f;
typedef __attribute__((ext_vector_type(8)))  unsigned int v8u;

#define IMG_H 512
#define IMG_W 512
#define N_IMG 96                    // 32 batch * 3 channels
// Each wave processes a 16-row x 64-col strip = 4 adjacent 16x16 tiles sharing
// one 26x74 input halo patch (loaded as 32x74, zero-padded).
#define GROUPS_PER_IMG 256          // 32 row-strips * 8 col-groups
#define TOTAL_GROUPS (N_IMG * GROUPS_PER_IMG)     // 24576
#define WAVES_PER_BLOCK 2
#define NBLOCKS (TOTAL_GROUPS / WAVES_PER_BLOCK)  // 12288
#define PATCH_COLS 74               // 64 + 2*5 halo
#define ROW_STRIDE 34               // 32 rows + pad (keeps b32 pairs aligned)
#define LOAD_ITEMS (16 * PATCH_COLS)  // row-pairs * cols = 1184

__device__ __forceinline__ unsigned short f2h(float f) {
    return __builtin_bit_cast(unsigned short, (_Float16)f);
}
__device__ __forceinline__ unsigned int pk(unsigned short lo, unsigned short hi) {
    return (unsigned int)lo | ((unsigned int)hi << 16);
}

__global__ __launch_bounds__(64)
void ssim_tiles_kernel(const float* __restrict__ img1,
                       const float* __restrict__ img2,
                       float* __restrict__ partial)
{
    // Per-wave private LDS: 5 signal patches stored TRANSPOSED [col][row] in f16
    // (even row pairs -> one aligned b32 per stage-1 B-operand register), plus a
    // 16x32 staging tile for the stage1(C-layout f32) -> stage2(A-layout f16)
    // lane/register transpose.
    __shared__ unsigned short sP[WAVES_PER_BLOCK][5][PATCH_COLS][ROW_STRIDE];
    __shared__ unsigned short sT[WAVES_PER_BLOCK][16][ROW_STRIDE];
    __shared__ float sSum[WAVES_PER_BLOCK];

    const int lane = threadIdx.x & 31;
    const int wv   = threadIdx.x >> 5;
    const int half = lane >> 4;      // 0: lanes 0-15, 1: lanes 16-31
    const int l15  = lane & 15;

    // ---- Gaussian taps: g (vertical pass) and wh = (g.g)*g (horizontal pass) ----
    float g[11];
    float z = 0.f;
    #pragma unroll
    for (int t = 0; t < 11; ++t) {
        float d = (float)(t - 5);
        g[t] = __expf(-(d * d) / 4.5f);   // 2*sigma^2 = 4.5
        z += g[t];
    }
    float invz = 1.f / z, gg = 0.f;
    #pragma unroll
    for (int t = 0; t < 11; ++t) { g[t] *= invz; gg += g[t] * g[t]; }
    float wh[11];
    #pragma unroll
    for (int t = 0; t < 11; ++t) wh[t] = g[t] * gg;

    // ---- Wv as WMMA A operand (16x32 f16): A[M,K] = g[K-M] if 0<=K-M<=10 ----
    // A-layout: lane M=l15; reg r -> K0 = 2r + (r>=4 ? 8:0) + half*8
    v8u wvU;
    #pragma unroll
    for (int r = 0; r < 8; ++r) {
        int K0 = 2 * r + (r >= 4 ? 8 : 0) + half * 8;
        float lo = 0.f, hi = 0.f;
        #pragma unroll
        for (int t = 0; t < 11; ++t) {
            if (K0     - l15 == t) lo = g[t];
            if (K0 + 1 - l15 == t) hi = g[t];
        }
        wvU[r] = pk(f2h(lo), f2h(hi));
    }
    const v16h WvA = __builtin_bit_cast(v16h, wvU);

    // ---- Wh as WMMA B operand (32x16 f16): B[K,N] = wh[K-N] if 0<=K-N<=10 ----
    // B-layout: lane N=l15; reg r -> K0 = 2r + (r>=4 ? 8:0) + half*8
    v8u whU;
    #pragma unroll
    for (int r = 0; r < 8; ++r) {
        int K0 = 2 * r + (r >= 4 ? 8 : 0) + half * 8;
        float lo = 0.f, hi = 0.f;
        #pragma unroll
        for (int t = 0; t < 11; ++t) {
            if (K0     - l15 == t) lo = wh[t];
            if (K0 + 1 - l15 == t) hi = wh[t];
        }
        whU[r] = pk(f2h(lo), f2h(hi));
    }
    const v16h WhB = __builtin_bit_cast(v16h, whU);

    // ---- Which 16x64 strip does this wave own? ----
    const int gw  = blockIdx.x * WAVES_PER_BLOCK + wv;    // 0..24575
    const int im  = gw >> 8;
    const int rem = gw & 255;
    const int R0  = (rem >> 3) << 4;    // strip top row
    const int C0g = (rem & 7) << 6;     // strip left col (64-wide)
    const float* xb = img1 + (size_t)im * (IMG_H * IMG_W);
    const float* yb = img2 + (size_t)im * (IMG_H * IMG_W);

    // Warm L2 for the start of this wave's patch (global_prefetch_b8 path).
    {
        int rr = R0 - 5 < 0 ? 0 : R0 - 5;
        int cc = C0g - 5 < 0 ? 0 : C0g - 5;
        __builtin_prefetch(xb + rr * IMG_W + cc, 0, 0);
        __builtin_prefetch(yb + rr * IMG_W + cc, 0, 0);
    }

    // ---- Load 32x74 patch once (zero-padded), form 5 signals, store f16 transposed ----
    for (int it = 0; it < (LOAD_ITEMS + 31) / 32; ++it) {
        int w = lane + it * 32;
        if (w < LOAD_ITEMS) {
            int rp  = w / PATCH_COLS;          // 0..15 row pair
            int col = w - rp * PATCH_COLS;     // 0..73
            int rr  = R0 - 5 + 2 * rp;
            int cc  = C0g - 5 + col;
            float x0 = 0.f, y0 = 0.f, x1 = 0.f, y1 = 0.f;
            bool cin = (cc >= 0) && (cc < IMG_W);
            if (cin && rr >= 0 && rr < IMG_H)         { x0 = xb[rr * IMG_W + cc];       y0 = yb[rr * IMG_W + cc]; }
            if (cin && rr + 1 >= 0 && rr + 1 < IMG_H) { x1 = xb[(rr + 1) * IMG_W + cc]; y1 = yb[(rr + 1) * IMG_W + cc]; }
            *(unsigned int*)&sP[wv][0][col][2 * rp] = pk(f2h(x0),      f2h(x1));
            *(unsigned int*)&sP[wv][1][col][2 * rp] = pk(f2h(y0),      f2h(y1));
            *(unsigned int*)&sP[wv][2][col][2 * rp] = pk(f2h(x0 * x0), f2h(x1 * x1));
            *(unsigned int*)&sP[wv][3][col][2 * rp] = pk(f2h(y0 * y0), f2h(y1 * y1));
            *(unsigned int*)&sP[wv][4][col][2 * rp] = pk(f2h(x0 * y0), f2h(x1 * y1));
        }
    }
    // DS ops from the same wave complete in order; all LDS regions are
    // wave-private, so no block barrier is required before consuming sP.

    float waveSum = 0.f;

    for (int tt = 0; tt < 4; ++tt) {          // 4 adjacent 16x16 tiles in the strip
        const int cb = tt * 16;               // patch column base of this tile

        v8f Out[5];
        #pragma unroll
        for (int s = 0; s < 5; ++s) {
            // Stage 1: T(16x32) = Wv(16x32) x P(32x32), two 16-col B blocks
            v8u blu, bru;
            #pragma unroll
            for (int r = 0; r < 8; ++r) {
                int K0 = 2 * r + (r >= 4 ? 8 : 0) + half * 8;
                blu[r] = *(const unsigned int*)&sP[wv][s][cb + l15][K0];
                bru[r] = *(const unsigned int*)&sP[wv][s][cb + l15 + 16][K0];
            }
            v16h Bl = __builtin_bit_cast(v16h, blu);
            v16h Br = __builtin_bit_cast(v16h, bru);
            v8f zero = {};
            v8f Tl = __builtin_amdgcn_wmma_f32_16x16x32_f16(false, WvA, false, Bl,
                                                            (short)0, zero, false, false);
            v8f Tr = __builtin_amdgcn_wmma_f32_16x16x32_f16(false, WvA, false, Br,
                                                            (short)0, zero, false, false);

            // Re-layout C(f32, M-in-regs/N-in-lanes) -> A(f16, M-in-lanes/K-in-regs)
            #pragma unroll
            for (int v = 0; v < 8; ++v) {
                int M = v + half * 8;
                sT[wv][M][l15]      = f2h(Tl[v]);
                sT[wv][M][l15 + 16] = f2h(Tr[v]);
            }
            v8u au;
            #pragma unroll
            for (int r = 0; r < 8; ++r) {
                int K0 = 2 * r + (r >= 4 ? 8 : 0) + half * 8;
                au[r] = *(const unsigned int*)&sT[wv][l15][K0];
            }
            v16h At = __builtin_bit_cast(v16h, au);

            // Stage 2: Out(16x16) = T(16x32) x Wh(32x16)
            Out[s] = __builtin_amdgcn_wmma_f32_16x16x32_f16(false, At, false, WhB,
                                                            (short)0, zero, false, false);
        }

        // ---- SSIM map (each lane owns 8 pixels in C-layout) ----
        // den >= C1*C2 > 0, so a fast hardware reciprocal (v_rcp_f32, ~2.5 ulp)
        // is more than accurate enough for the mean; avoids the precise-div
        // v_div_scale/fix-up expansion (~10 VALU/pixel).
        #pragma unroll
        for (int v = 0; v < 8; ++v) {
            float mu1 = Out[0][v], mu2 = Out[1][v];
            float s1  = Out[2][v] - mu1 * mu1;
            float s2  = Out[3][v] - mu2 * mu2;
            float s12 = Out[4][v] - mu1 * mu2;
            float num = (2.f * mu1 * mu2 + 1e-4f) * (2.f * s12 + 9e-4f);
            float den = (mu1 * mu1 + mu2 * mu2 + 1e-4f) * (s1 + s2 + 9e-4f);
            waveSum += num * __builtin_amdgcn_rcpf(den);
        }
    }

    // ---- deterministic reduction: wave shuffle -> fixed-order block sum ----
    #pragma unroll
    for (int off = 16; off >= 1; off >>= 1)
        waveSum += __shfl_xor(waveSum, off, 32);
    if (lane == 0) sSum[wv] = waveSum;
    __syncthreads();
    if (threadIdx.x == 0) {
        float b = 0.f;
        #pragma unroll
        for (int i = 0; i < WAVES_PER_BLOCK; ++i) b += sSum[i];
        partial[blockIdx.x] = b;
    }
}

__global__ __launch_bounds__(256)
void ssim_finalize_kernel(const float* __restrict__ partial, float* __restrict__ out,
                          int n, float invN)
{
    __shared__ float red[256];
    float a = 0.f;
    for (int i = threadIdx.x; i < n; i += 256) a += partial[i];  // fixed order per thread
    red[threadIdx.x] = a;
    __syncthreads();
    for (int s = 128; s > 0; s >>= 1) {
        if ((int)threadIdx.x < s) red[threadIdx.x] += red[threadIdx.x + s];
        __syncthreads();
    }
    if (threadIdx.x == 0) out[0] = 1.0f - red[0] * invN;
}

extern "C" void kernel_launch(void* const* d_in, const int* in_sizes, int n_in,
                              void* d_out, int out_size, void* d_ws, size_t ws_size,
                              hipStream_t stream) {
    const float* img1 = (const float*)d_in[0];
    const float* img2 = (const float*)d_in[1];
    float* partial = (float*)d_ws;          // NBLOCKS floats (48 KB)
    float* out = (float*)d_out;

    ssim_tiles_kernel<<<NBLOCKS, 64, 0, stream>>>(img1, img2, partial);
    ssim_finalize_kernel<<<1, 256, 0, stream>>>(partial, out, NBLOCKS,
                                                1.0f / 25165824.0f);
}
